// MultiHeadAttention1D_39994735461022
// MI455X (gfx1250) — matrix operand order repaired and checked by reading, not compiled, not run on hardware
//
#include <hip/hip_runtime.h>
#include <hip/hip_bf16.h>
#include <math.h>
#include <stdint.h>

// ---------------------------------------------------------------------------
// MHA (B=2, S=2048, H=2048, NH=16, HD=128), causal, bf16 WMMA pipeline:
//   1) convert hs -> bf16, transpose+convert w_qkv / w_out -> [N,K] bf16
//   2) GEMM1: qkv(bf16) = hs @ w_qkv + b_qkv         (WMMA bf16, f32 acc)
//   3) flash attention per (b,h,qtile)               (WMMA bf16, f32 acc)
//   4) GEMM2: out(f32) = ctx @ w_out + b_out         (WMMA bf16, f32 acc)
// Tile staging uses the CDNA5 Tensor Data Mover (tensor_load_to_lds +
// s_wait_tensorcnt) for all non-transposed 2D tiles; VGPR-path fallback kept.
// ---------------------------------------------------------------------------

typedef __bf16 bf16;
typedef __attribute__((ext_vector_type(16))) __bf16 bf16x16;
typedef __attribute__((ext_vector_type(8)))  __bf16 bf16x8;
typedef __attribute__((ext_vector_type(8)))  float  f32x8;
typedef __attribute__((ext_vector_type(4)))  unsigned int u32x4;
typedef __attribute__((ext_vector_type(8)))  int    i32x8;
typedef __attribute__((ext_vector_type(4)))  int    i32x4;

#define H_DIM  2048
#define NHEAD  16
#define HDIM   128
#define SEQ    2048
#define BATCH  2
#define MTOT   (BATCH * SEQ)   // 4096
#define QKV_N  (3 * H_DIM)     // 6144

#if defined(__gfx1250__) && __has_builtin(__builtin_amdgcn_tensor_load_to_lds)
#define USE_TDM 1
#else
#define USE_TDM 0
#endif

#if USE_TDM
// 2D TDM load: tile_d1 rows x tile_d0 bf16 elems, global row stride
// `stride_elems`, landing contiguously (row-major) at LDS byte offset lds_off.
// D# layout per CDNA5 ISA 8.3/8.4 (group0 128b, group1 256b; groups 2/3 zero).
static __device__ __forceinline__ void
tdm_load_2d_bf16(unsigned lds_off, const void* gptr,
                 int tile_d0, int tile_d1, long stride_elems)
{
    const unsigned long long ga = (unsigned long long)(uintptr_t)gptr;
    const unsigned td0 = (unsigned)tile_d0;   // tensor_dim0 == tile_dim0
    const unsigned td1 = (unsigned)tile_d1;   // tensor_dim1 == tile_dim1
    const unsigned long long s0 = (unsigned long long)stride_elems;

    u32x4 g0;
    g0[0] = 1u;                                            // count=1, user D#
    g0[1] = lds_off;                                       // lds_addr (bytes)
    g0[2] = (unsigned)(ga & 0xFFFFFFFFull);                // global_addr[31:0]
    g0[3] = (unsigned)((ga >> 32) & 0x01FFFFFFull)         // global_addr[56:32]
          | 0x80000000u;                                   // type=2 (image)

    i32x8 g1;
    g1[0] = (int)(1u << 16);                               // data_size=1 (2B)
    g1[1] = (int)((td0 & 0xFFFFu) << 16);                  // tensor_dim0 lo
    g1[2] = (int)((td0 >> 16) | ((td1 & 0xFFFFu) << 16));  // dim0 hi | dim1 lo
    g1[3] = (int)((td1 >> 16) | ((unsigned)tile_d0 << 16));// dim1 hi | tile_d0
    g1[4] = (int)((unsigned)tile_d1 & 0xFFFFu);            // tile_d1, tile_d2=0
    g1[5] = (int)(unsigned)(s0 & 0xFFFFFFFFull);           // dim0_stride[31:0]
    g1[6] = (int)(unsigned)((s0 >> 32) & 0xFFFFull);       // dim0_stride[47:32]
    g1[7] = 0;

    const i32x4 z4 = {0, 0, 0, 0};
#if __clang_major__ >= 23
    const i32x8 z8 = {0, 0, 0, 0, 0, 0, 0, 0};
    __builtin_amdgcn_tensor_load_to_lds(g0, g1, z4, z4, z8, 0);
#else
    __builtin_amdgcn_tensor_load_to_lds(g0, g1, z4, z4, 0);
#endif
}
#endif // USE_TDM

// A-operand fragment (16x32 bf16 tile, row-major in LDS, stride in elems).
// ISA layout: lanes 0-15 -> M=lane, K={0..7,16..23}; lanes 16-31 -> K={8..15,24..31}
static __device__ __forceinline__ bf16x16
make_fragA(const bf16* base, int stride, int lane) {
    const int r  = lane & 15;
    const int kb = (lane >> 4) << 3;          // 0 or 8
    const bf16* p = base + r * stride + kb;
    bf16x8 lo = *(const bf16x8*)(p);
    bf16x8 hi = *(const bf16x8*)(p + 16);
    return __builtin_shufflevector(lo, hi, 0,1,2,3,4,5,6,7,8,9,10,11,12,13,14,15);
}

// B-operand fragment (32x16 bf16, stored N-major in LDS: row n holds K contig).
// ISA layout: lanes 0-15 -> N=lane, K={0..7,16..23}; lanes 16-31 -> N=lane-16, K={8..15,24..31}
static __device__ __forceinline__ bf16x16
make_fragB(const bf16* base, int stride, int lane) {
    const int n  = lane & 15;
    const int kb = (lane >> 4) << 3;          // 0 or 8
    const bf16* p = base + n * stride + kb;
    bf16x8 lo = *(const bf16x8*)(p);
    bf16x8 hi = *(const bf16x8*)(p + 16);
    return __builtin_shufflevector(lo, hi, 0,1,2,3,4,5,6,7,8,9,10,11,12,13,14,15);
}

// ---------------------------------------------------------------------------
// Generic tiled WMMA GEMM:  C[M,N] = A[M,K](bf16) * BT[N,K](bf16)^T + bias[N]
// Block tile 128x128, K-stage 64, 8 waves (2x4), wave tile 64x32 (4x2 WMMA).
// Tiles staged by TDM (wave 0 issues; TENSORcnt + barrier publishes to WG).
// ---------------------------------------------------------------------------
template <typename OutT>
__global__ __launch_bounds__(256)
void gemm_bf16_wmma(const bf16* __restrict__ A,
                    const bf16* __restrict__ BT,
                    const float* __restrict__ bias,
                    OutT* __restrict__ C,
                    int M, int N, int K)
{
    constexpr int TM = 128, TN = 128, TK = 64;
    __shared__ __align__(16) bf16 As[TM][TK];   // 16 KB
    __shared__ __align__(16) bf16 Bs[TN][TK];   // 16 KB

    const int tid  = threadIdx.x;
    const int lane = tid & 31;
    const int wave = tid >> 5;      // 0..7
    const int wm   = wave >> 2;     // 0..1 -> 64-row slab
    const int wn   = wave & 3;      // 0..3 -> 32-col slab
    const int m0   = blockIdx.y * TM;
    const int n0   = blockIdx.x * TN;

    const f32x8 zero = {0.f,0.f,0.f,0.f,0.f,0.f,0.f,0.f};
    f32x8 acc[4][2];
    #pragma unroll
    for (int i = 0; i < 4; ++i)
        #pragma unroll
        for (int j = 0; j < 2; ++j) acc[i][j] = zero;

    for (int k0 = 0; k0 < K; k0 += TK) {
#if USE_TDM
        // previous end-of-loop barrier guarantees the WG is done reading As/Bs
        if (wave == 0) {
            tdm_load_2d_bf16((unsigned)(size_t)&As[0][0],
                             &A[(long)m0 * K + k0], TK, TM, K);
            tdm_load_2d_bf16((unsigned)(size_t)&Bs[0][0],
                             &BT[(long)n0 * K + k0], TK, TN, K);
            __builtin_amdgcn_s_wait_tensorcnt(0);
        }
#else
        #pragma unroll
        for (int i = 0; i < (TM * TK) / (256 * 8); ++i) {
            int idx = (i * 256 + tid) * 8;
            int r = idx / TK, c = idx % TK;
            *(bf16x8*)&As[r][c] = *(const bf16x8*)&A[(long)(m0 + r) * K + k0 + c];
        }
        #pragma unroll
        for (int i = 0; i < (TN * TK) / (256 * 8); ++i) {
            int idx = (i * 256 + tid) * 8;
            int r = idx / TK, c = idx % TK;
            *(bf16x8*)&Bs[r][c] = *(const bf16x8*)&BT[(long)(n0 + r) * K + k0 + c];
        }
#endif
        __syncthreads();

        #pragma unroll
        for (int kk = 0; kk < TK; kk += 32) {
            bf16x16 af[4], bfr[2];
            #pragma unroll
            for (int i = 0; i < 4; ++i)
                af[i] = make_fragA(&As[wm * 64 + i * 16][kk], TK, lane);
            #pragma unroll
            for (int j = 0; j < 2; ++j)
                bfr[j] = make_fragB(&Bs[wn * 32 + j * 16][kk], TK, lane);
            #pragma unroll
            for (int i = 0; i < 4; ++i)
                #pragma unroll
                for (int j = 0; j < 2; ++j)
                    acc[i][j] = __builtin_amdgcn_wmma_f32_16x16x32_bf16(
                        false, af[i], false, bfr[j], (short)0, acc[i][j], false, false);
        }
        __syncthreads();
    }

    // epilogue: C/D layout -> lane owns col n=(lane&15); reg r -> row r+8*(lane>=16)
    const int hl = lane >> 4;
    #pragma unroll
    for (int i = 0; i < 4; ++i) {
        #pragma unroll
        for (int j = 0; j < 2; ++j) {
            int nn = n0 + wn * 32 + j * 16 + (lane & 15);
            float bv = bias[nn];
            #pragma unroll
            for (int r = 0; r < 8; ++r) {
                int mm = m0 + wm * 64 + i * 16 + r + (hl << 3);
                C[(long)mm * N + nn] = (OutT)(acc[i][j][r] + bv);
            }
        }
    }
}

// ---------------------------------------------------------------------------
// Flash attention: grid (S/64, B*NH), 4 waves. Wave owns 16 q rows, full HD=128.
// qkv row layout: head h -> columns [h*384, h*384+384) as [Q(128)|K(128)|V(128)]
// Q/K tiles staged by TDM; V staged manually (needs transpose into d-major).
// ---------------------------------------------------------------------------
__global__ __launch_bounds__(128)
void attn_flash_wmma(const bf16* __restrict__ qkv,   // [MTOT, 6144] bf16
                     bf16* __restrict__ ctx)         // [MTOT, 2048] bf16
{
    __shared__ __align__(16) bf16 Qs[64][HDIM];      // 16 KB
    __shared__ __align__(16) bf16 Ks[64][HDIM];      // 16 KB
    __shared__ __align__(16) bf16 Vt[HDIM][64];      // 16 KB (d-major)
    __shared__ __align__(16) bf16 Ps[4][16][64];     //  8 KB (per-wave probs)

    const int tid  = threadIdx.x;
    const int lane = tid & 31;
    const int wave = tid >> 5;                 // 0..3
    const int hl   = lane >> 4;                // lane half
    const int bh   = blockIdx.y;               // 0..31
    const int b    = bh >> 4;
    const int h    = bh & 15;
    const int qb   = blockIdx.x * 64;

    const long rs = QKV_N;
    const bf16* Qg = qkv + (long)(b * SEQ) * QKV_N + h * (3 * HDIM);
    const bf16* Kg = Qg + HDIM;
    const bf16* Vg = Qg + 2 * HDIM;

    // stage Q tile (64 x 128)
#if USE_TDM
    if (wave == 0)
        tdm_load_2d_bf16((unsigned)(size_t)&Qs[0][0],
                         &Qg[(long)qb * rs], HDIM, 64, rs);
#else
    #pragma unroll
    for (int i = 0; i < (64 * HDIM) / (128 * 8); ++i) {
        int idx = (i * 128 + tid) * 8;
        int r = idx / HDIM, c = idx % HDIM;
        *(bf16x8*)&Qs[r][c] = *(const bf16x8*)&Qg[(long)(qb + r) * rs + c];
    }
#endif

    const f32x8 zero = {0.f,0.f,0.f,0.f,0.f,0.f,0.f,0.f};
    f32x8 acc[8];
    #pragma unroll
    for (int d = 0; d < 8; ++d) acc[d] = zero;
    float run_max[8], run_sum[8];
    #pragma unroll
    for (int r = 0; r < 8; ++r) { run_max[r] = -3e30f; run_sum[r] = 0.f; }

    const float scale = 0.08838834764831845f;  // 1/sqrt(128)

    for (int j = 0; j <= (int)blockIdx.x; ++j) {
        const int kb0 = j * 64;
        // stage K tile (TDM) and transposed V tile (manual)
#if USE_TDM
        if (wave == 0)
            tdm_load_2d_bf16((unsigned)(size_t)&Ks[0][0],
                             &Kg[(long)kb0 * rs], HDIM, 64, rs);
        #pragma unroll
        for (int i = 0; i < 8; ++i) {
            int idx = (i * 128 + tid) * 8;
            int r = idx / HDIM, c = idx % HDIM;
            bf16x8 v = *(const bf16x8*)&Vg[(long)(kb0 + r) * rs + c];
            #pragma unroll
            for (int e = 0; e < 8; ++e) Vt[c + e][r] = v[e];
        }
        if (wave == 0) __builtin_amdgcn_s_wait_tensorcnt(0);
#else
        #pragma unroll
        for (int i = 0; i < 8; ++i) {
            int idx = (i * 128 + tid) * 8;
            int r = idx / HDIM, c = idx % HDIM;
            *(bf16x8*)&Ks[r][c] = *(const bf16x8*)&Kg[(long)(kb0 + r) * rs + c];
            bf16x8 v = *(const bf16x8*)&Vg[(long)(kb0 + r) * rs + c];
            #pragma unroll
            for (int e = 0; e < 8; ++e) Vt[c + e][r] = v[e];
        }
#endif
        // prefetch next K/V tiles into GL2 while this one computes
        if (j < (int)blockIdx.x) {
            const bf16* nk = &Kg[(long)(kb0 + 64 + (tid & 63)) * rs];
            const bf16* nv = &Vg[(long)(kb0 + 64 + (tid & 63)) * rs];
            __builtin_prefetch(nk, 0, 1);
            __builtin_prefetch(nv, 0, 1);
        }
        __syncthreads();

        // scores S = Q K^T (16x64 per wave): 16 WMMA
        f32x8 s[4];
        #pragma unroll
        for (int nt = 0; nt < 4; ++nt) s[nt] = zero;
        #pragma unroll
        for (int kk = 0; kk < HDIM; kk += 32) {
            bf16x16 aq = make_fragA(&Qs[wave * 16][kk], HDIM, lane);
            #pragma unroll
            for (int nt = 0; nt < 4; ++nt) {
                bf16x16 bk = make_fragB(&Ks[nt * 16][kk], HDIM, lane);
                s[nt] = __builtin_amdgcn_wmma_f32_16x16x32_bf16(
                    false, aq, false, bk, (short)0, s[nt], false, false);
            }
        }

        // scale + causal mask (diagonal tile only) + online softmax
        const bool diag = (j == (int)blockIdx.x);
        #pragma unroll
        for (int r = 0; r < 8; ++r) {
            const int qrow = qb + wave * 16 + r + (hl << 3);
            float mv = -3e30f;
            #pragma unroll
            for (int nt = 0; nt < 4; ++nt) {
                float sv = s[nt][r] * scale;
                int kcol = kb0 + nt * 16 + (lane & 15);
                if (diag && kcol > qrow) sv = -10000.0f;
                s[nt][r] = sv;
                mv = fmaxf(mv, sv);
            }
            // row-max across the 16 lanes holding this row (xor<16 stays in half)
            #pragma unroll
            for (int off = 1; off < 16; off <<= 1)
                mv = fmaxf(mv, __shfl_xor(mv, off));
            float nm   = fmaxf(run_max[r], mv);
            float corr = __expf(run_max[r] - nm);
            run_max[r] = nm;
            float rsum = 0.f;
            #pragma unroll
            for (int nt = 0; nt < 4; ++nt) {
                float p = __expf(s[nt][r] - nm);
                s[nt][r] = p;
                rsum += p;
            }
            #pragma unroll
            for (int off = 1; off < 16; off <<= 1)
                rsum += __shfl_xor(rsum, off);
            run_sum[r] = run_sum[r] * corr + rsum;
            #pragma unroll
            for (int d = 0; d < 8; ++d) acc[d][r] *= corr;
        }

        // P (C-layout) -> LDS -> A-layout fragments
        #pragma unroll
        for (int nt = 0; nt < 4; ++nt) {
            int ncol = nt * 16 + (lane & 15);
            #pragma unroll
            for (int r = 0; r < 8; ++r)
                Ps[wave][r + (hl << 3)][ncol] = (bf16)s[nt][r];
        }
        __syncthreads();

        // ctx += P @ V : 16 WMMA (8 d-tiles x 2 k-steps)
        #pragma unroll
        for (int kk = 0; kk < 64; kk += 32) {
            bf16x16 ap = make_fragA(&Ps[wave][0][kk], 64, lane);
            #pragma unroll
            for (int dt = 0; dt < 8; ++dt) {
                bf16x16 bv = make_fragB(&Vt[dt * 16][kk], 64, lane);
                acc[dt] = __builtin_amdgcn_wmma_f32_16x16x32_bf16(
                    false, ap, false, bv, (short)0, acc[dt], false, false);
            }
        }
        __syncthreads();
    }

    // normalize and store ctx (head h -> cols [h*128, h*128+128))
    #pragma unroll
    for (int dt = 0; dt < 8; ++dt) {
        int dcol = h * HDIM + dt * 16 + (lane & 15);
        #pragma unroll
        for (int r = 0; r < 8; ++r) {
            int srow = qb + wave * 16 + r + (hl << 3);
            float v = acc[dt][r] / run_sum[r];
            ctx[(long)(b * SEQ + srow) * H_DIM + dcol] = (bf16)v;
        }
    }
}

// ---------------------------------------------------------------------------
// conversions
// ---------------------------------------------------------------------------
__global__ __launch_bounds__(256)
void convert_f32_bf16(const float* __restrict__ x, bf16* __restrict__ y, long n)
{
    long i = ((long)blockIdx.x * blockDim.x + threadIdx.x) * 4;
    if (i + 3 < n) {
        #pragma unroll
        for (int e = 0; e < 4; ++e) y[i + e] = (bf16)x[i + e];
    } else {
        for (long k = i; k < n; ++k) y[k] = (bf16)x[k];
    }
}

__global__ __launch_bounds__(256)
void transpose_f32_to_bf16(const float* __restrict__ W,  // [K,N] f32
                           bf16* __restrict__ WT,        // [N,K] bf16
                           int K, int N)
{
    __shared__ float tile[32][33];
    const int kb = blockIdx.y * 32;
    const int nb = blockIdx.x * 32;
    const int tx = threadIdx.x & 31;
    const int ty = threadIdx.x >> 5;   // 0..7
    #pragma unroll
    for (int i = ty; i < 32; i += 8)
        tile[i][tx] = W[(long)(kb + i) * N + nb + tx];
    __syncthreads();
    #pragma unroll
    for (int i = ty; i < 32; i += 8)
        WT[(long)(nb + i) * K + kb + tx] = (bf16)tile[tx][i];
}

// ---------------------------------------------------------------------------
extern "C" void kernel_launch(void* const* d_in, const int* in_sizes, int n_in,
                              void* d_out, int out_size, void* d_ws, size_t ws_size,
                              hipStream_t stream)
{
    (void)in_sizes; (void)n_in; (void)out_size; (void)ws_size;
    const float* hs    = (const float*)d_in[0];  // [2,2048,2048]
    const float* w_qkv = (const float*)d_in[1];  // [2048,6144]
    const float* b_qkv = (const float*)d_in[2];  // [6144]
    const float* w_out = (const float*)d_in[3];  // [2048,2048]
    const float* b_out = (const float*)d_in[4];  // [2048]
    float* out = (float*)d_out;                  // [2,2048,2048]

    char* ws = (char*)d_ws;
    bf16* hsb   = (bf16*)ws;  ws += (size_t)MTOT  * H_DIM * sizeof(bf16);  // 16 MB
    bf16* wqkvT = (bf16*)ws;  ws += (size_t)QKV_N * H_DIM * sizeof(bf16);  // 24 MB
    bf16* woutT = (bf16*)ws;  ws += (size_t)H_DIM * H_DIM * sizeof(bf16);  //  8 MB
    bf16* qkv   = (bf16*)ws;  ws += (size_t)MTOT  * QKV_N * sizeof(bf16);  // 48 MB
    bf16* ctx   = (bf16*)ws;                                               // 16 MB

    {
        long n = (long)MTOT * H_DIM;
        long blocks = (n / 4 + 255) / 256;
        convert_f32_bf16<<<dim3((unsigned)blocks), dim3(256), 0, stream>>>(hs, hsb, n);
    }
    transpose_f32_to_bf16<<<dim3(QKV_N / 32, H_DIM / 32), dim3(256), 0, stream>>>(
        w_qkv, wqkvT, H_DIM, QKV_N);
    transpose_f32_to_bf16<<<dim3(H_DIM / 32, H_DIM / 32), dim3(256), 0, stream>>>(
        w_out, woutT, H_DIM, H_DIM);

    gemm_bf16_wmma<bf16><<<dim3(QKV_N / 128, MTOT / 128), dim3(256), 0, stream>>>(
        hsb, wqkvT, b_qkv, qkv, MTOT, QKV_N, H_DIM);

    attn_flash_wmma<<<dim3(SEQ / 64, BATCH * NHEAD), dim3(128), 0, stream>>>(qkv, ctx);

    gemm_bf16_wmma<float><<<dim3(H_DIM / 128, MTOT / 128), dim3(256), 0, stream>>>(
        ctx, woutT, b_out, out, MTOT, H_DIM, H_DIM);
}
